// Mask_43843026157955
// MI455X (gfx1250) — compile-verified, matrix operand order checked by reading, not run on hardware
//
#include <hip/hip_runtime.h>

// ---------------------------------------------------------------------------
// MI455X (gfx1250, wave32) implementation, round 3.
//
//   1) nchw_to_nhwc_f16 : x f32 NCHW -> xb f16 NHWC via 32x32 LDS transpose
//   2) pack_w           : w[co][ci][3][3] f32 -> wp[rs][ci][co] f16
//   3) conv_wmma        : implicit-GEMM conv + bias + ReLU on
//                         v_wmma_f32_16x16x32_f16. Loop order: k0 outer,
//                         taps inner -> X tile staged ONCE per k0 via
//                         global_load_async_to_lds_b128 (ASYNCcnt-tracked),
//                         72 WMMAs per barrier pair.
//   4) conv3_topk_mask  : 1x1 conv + ReLU + per-image top-5 -> 16x16 mask
//   5) upsample_kernel  : 32x nearest upsample, float4 streaming stores
// ---------------------------------------------------------------------------

typedef __attribute__((ext_vector_type(16))) _Float16 v16h;
typedef __attribute__((ext_vector_type(8)))  _Float16 v8h;
typedef __attribute__((ext_vector_type(8)))  float    v8f;
typedef int v4i __attribute__((vector_size(16)));

#if defined(__gfx1250__) && \
    __has_builtin(__builtin_amdgcn_global_load_async_to_lds_b128) && \
    __has_builtin(__builtin_amdgcn_s_wait_asynccnt)
#define USE_ASYNC_LDS 1
// Pointer-to-v4i in global (AS1) / LDS (AS3) address spaces, per the builtin's
// signature reported by hipcc: "int __vector(4) __device__ *".
typedef __attribute__((address_space(1))) v4i* async_gptr;
typedef __attribute__((address_space(3))) v4i* async_lptr;
#endif

// CDNA5 16-bit 16x32 operand fragment load from LDS rows of 32 halfs:
//   lane L<16 : row=L,    VGPRs 0-3 K=0..7,   VGPRs 4-7 K=16..23
//   lane L>=16: row=L-16, VGPRs 0-3 K=8..15,  VGPRs 4-7 K=24..31
// -> two ds_load_b128 per fragment per lane.
__device__ __forceinline__ v16h load_frag(const _Float16* base, int lane) {
  const int row = lane & 15;
  const int hi  = (lane >> 4) & 1;
  const _Float16* r = base + row * 32;
  union { v16h v; v8h h[2]; } u;
  u.h[0] = *(const v8h*)(r + hi * 8);
  u.h[1] = *(const v8h*)(r + 16 + hi * 8);
  return u.v;
}

// A-fragment with im2col shift applied at load time from the un-shifted
// X tile (Xlds[pixel][32ci]); out-of-image rows become zero (pad=1).
__device__ __forceinline__ v16h load_a_frag(const _Float16* Xlds, int mbase,
                                            int lane, int dy, int dx) {
  const int row = mbase + (lane & 15);
  const int py  = (row >> 4) + dy;
  const int px  = (row & 15) + dx;
  const int hi  = (lane >> 4) & 1;
  union { v16h v; v8h h[2]; } u;
  if (((unsigned)py < 16u) && ((unsigned)px < 16u)) {
    const _Float16* r = Xlds + (py * 16 + px) * 32;
    u.h[0] = *(const v8h*)(r + hi * 8);
    u.h[1] = *(const v8h*)(r + 16 + hi * 8);
  } else {
    u.h[0] = (v8h)(_Float16)0.0f;
    u.h[1] = (v8h)(_Float16)0.0f;
  }
  return u.v;
}

// x f32 [B][C][16][16] -> xb f16 [B][256][C] (coalesced both sides via LDS).
__global__ __launch_bounds__(256)
void nchw_to_nhwc_f16(const float* __restrict__ in, _Float16* __restrict__ out, int C) {
  __shared__ float tile[32][33];
  const int b  = blockIdx.z;
  const int c0 = blockIdx.y * 32;
  const int p0 = blockIdx.x * 32;
  const int tx = threadIdx.x;       // 0..31
  const int ty = threadIdx.y;       // 0..7
#pragma unroll
  for (int r = 0; r < 32; r += 8)   // read: ci rows, p contiguous
    tile[ty + r][tx] = in[((size_t)b * C + (c0 + ty + r)) * 256 + p0 + tx];
  __syncthreads();
#pragma unroll
  for (int r = 0; r < 32; r += 8)   // write: p rows, ci contiguous
    out[((size_t)b * 256 + (p0 + ty + r)) * C + c0 + tx] = (_Float16)tile[tx][ty + r];
}

// w[co][ci][3][3] f32 -> wp[rs][ci][co] f16 (contiguous in co = GEMM N dim)
__global__ void pack_w(const float* __restrict__ w, _Float16* __restrict__ wp,
                       int Cout, int Cin) {
  int i = blockIdx.x * blockDim.x + threadIdx.x;
  int total = Cout * Cin * 9;
  if (i >= total) return;
  int co  = i / (Cin * 9);
  int rem = i % (Cin * 9);
  int ci  = rem / 9;
  int rs  = rem % 9;
  wp[((size_t)rs * Cin + ci) * Cout + co] = (_Float16)w[i];
}

// Implicit-GEMM 3x3 conv (pad=1) + bias + ReLU.
// grid = (Cout/64, B), block = 256 (8 waves, 4(M)x2(N) wave grid).
// Per k0: stage X tile (256px x 32ci, async-to-LDS if available) + 9 weight
// tiles, then 9 taps x (4x2) WMMAs = 72 v_wmma per barrier pair.
__global__ __launch_bounds__(256)
void conv_wmma(const _Float16* __restrict__ xin,    // [B][256][Cin] f16 NHWC
               const _Float16* __restrict__ wp,     // [9][Cin][Cout] f16
               const float*    __restrict__ bias,   // [Cout] f32
               _Float16*       __restrict__ outact, // [B][256][Cout] f16 NHWC
               int Cin, int Cout) {
  __shared__ _Float16 Xlds[256 * 32];      // 16 KB
  __shared__ _Float16 Blds[9 * 64 * 32];   // 36 KB, per-tap col-major [n*32+k]

  const int b      = blockIdx.y;
  const int co_blk = blockIdx.x * 64;
  const int tid    = threadIdx.x;
  const int lane   = tid & 31;
  const int wave   = tid >> 5;
  const int m0     = (wave >> 1) * 64;  // pixel-tile base for this wave
  const int n0     = (wave & 1) * 32;   // co-tile base (within block)

  v8f acc[4][2];
#pragma unroll
  for (int i = 0; i < 4; ++i)
#pragma unroll
    for (int j = 0; j < 2; ++j) acc[i][j] = (v8f)(0.0f);

  const int p  = tid;                    // pixel this thread stages
  const int kk = tid >> 3;               // 0..31 : ci row for B staging
  const int nb = (tid & 7) * 8;          // 0..56 : co group for B staging

  for (int k0 = 0; k0 < Cin; k0 += 32) {
    // --- stage X tile: 32 contiguous halfs per pixel (4 x 16B) ---
    {
      const _Float16* src = xin + ((size_t)b * 256 + p) * Cin + k0;
#ifdef USE_ASYNC_LDS
#pragma unroll
      for (int j = 0; j < 4; ++j)
        __builtin_amdgcn_global_load_async_to_lds_b128(
            (async_gptr)(src + j * 8),
            (async_lptr)(Xlds + p * 32 + j * 8),
            0, 0);
#else
      const v8h* s8 = (const v8h*)src;
      v8h* d8 = (v8h*)(Xlds + p * 32);
#pragma unroll
      for (int j = 0; j < 4; ++j) d8[j] = s8[j];
#endif
    }
    // --- stage 9 weight tiles: 16B contiguous load, col-major scatter ---
#pragma unroll
    for (int rs = 0; rs < 9; ++rs) {
      const v8h wv = *(const v8h*)(wp + ((size_t)(rs * Cin + k0 + kk)) * Cout + co_blk + nb);
      _Float16* dst = Blds + rs * 2048 + nb * 32 + kk;
#pragma unroll
      for (int j = 0; j < 8; ++j) dst[j * 32] = wv[j];
    }
#ifdef USE_ASYNC_LDS
    __builtin_amdgcn_s_wait_asynccnt(0);
#endif
    __syncthreads();

    // --- 9 taps x 4x2 WMMA ---
    for (int rs = 0; rs < 9; ++rs) {
      const int dy = rs / 3 - 1, dx = rs % 3 - 1;
      v16h a[4], bb[2];
#pragma unroll
      for (int i = 0; i < 4; ++i)
        a[i] = load_a_frag(Xlds, m0 + i * 16, lane, dy, dx);
#pragma unroll
      for (int j = 0; j < 2; ++j)
        bb[j] = load_frag(Blds + rs * 2048 + (n0 + j * 16) * 32, lane);
#pragma unroll
      for (int i = 0; i < 4; ++i)
#pragma unroll
        for (int j = 0; j < 2; ++j)
          acc[i][j] = __builtin_amdgcn_wmma_f32_16x16x32_f16(
              false, a[i], false, bb[j], (short)0, acc[i][j], false, false);
    }
    __syncthreads();
  }

  // Epilogue: bias + ReLU, store f16 NHWC [B][256][Cout].
  // C/D layout: col N = lane&15, row M = vgpr + 8*(lane>=16).
  const int colL  = lane & 15;
  const int rowHi = (lane >> 4) * 8;
#pragma unroll
  for (int i = 0; i < 4; ++i) {
#pragma unroll
    for (int j = 0; j < 2; ++j) {
      const int co = co_blk + n0 + j * 16 + colL;
      const float bv = bias[co];
      union { v8f v; float f[8]; } u;
      u.v = acc[i][j];
#pragma unroll
      for (int v = 0; v < 8; ++v) {
        const int pm = m0 + i * 16 + rowHi + v;
        float val = u.f[v] + bv;
        val = val > 0.0f ? val : 0.0f;
        outact[((size_t)b * 256 + pm) * Cout + co] = (_Float16)val;
      }
    }
  }
}

// 1x1 conv (128ch, NHWC -> contiguous b128 reads) + bias + ReLU,
// per-image top-5 -> binary mask [B][256].
__global__ __launch_bounds__(256)
void conv3_topk_mask(const _Float16* __restrict__ act2, const float* __restrict__ w3,
                     const float* __restrict__ b3, float* __restrict__ mask) {
  __shared__ float w3s[128];
  __shared__ float sc[256];
  __shared__ float rv[256];
  __shared__ int   ri[256];
  __shared__ unsigned char sel[256];

  const int b = blockIdx.x;
  const int p = threadIdx.x;
  if (p < 128) w3s[p] = w3[p];
  sel[p] = 0;
  __syncthreads();

  float s = 0.0f;
  const v8h* av = (const v8h*)(act2 + ((size_t)b * 256 + p) * 128);
#pragma unroll
  for (int c8 = 0; c8 < 16; ++c8) {
    const v8h hv = av[c8];
#pragma unroll
    for (int j = 0; j < 8; ++j) s += (float)hv[j] * w3s[c8 * 8 + j];
  }
  float score = s + b3[0];
  score = score > 0.0f ? score : 0.0f;
  sc[p] = score;
  __syncthreads();

  for (int it = 0; it < 5; ++it) {
    rv[p] = sc[p];
    ri[p] = p;
    __syncthreads();
    for (int st = 128; st > 0; st >>= 1) {
      if (p < st) {
        // strict '>' keeps the lower index on ties (matches lax.top_k)
        if (rv[p + st] > rv[p]) { rv[p] = rv[p + st]; ri[p] = ri[p + st]; }
      }
      __syncthreads();
    }
    if (p == 0) { sel[ri[0]] = 1; sc[ri[0]] = -1.0f; }
    __syncthreads();
  }

  mask[b * 256 + p] = (sel[p] && score > 0.0f) ? 1.0f : 0.0f;
}

// 32x nearest-neighbor upsample: out[b][0][Y][X] = mask[b][(Y/32)*16 + X/32].
__global__ void upsample_kernel(const float* __restrict__ mask, float* __restrict__ out) {
  const size_t t   = (size_t)blockIdx.x * blockDim.x + threadIdx.x;
  const size_t idx = t * 4;                        // 4 floats; cell const across vec
  const int b   = (int)(idx >> 18);                // 512*512 = 262144
  const int rem = (int)(idx & 262143);
  const int Y = rem >> 9;
  const int X = rem & 511;
  const float v = mask[b * 256 + ((Y >> 5) << 4) + (X >> 5)];
  *(float4*)(out + idx) = make_float4(v, v, v, v);
}

extern "C" void kernel_launch(void* const* d_in, const int* in_sizes, int n_in,
                              void* d_out, int out_size, void* d_ws, size_t ws_size,
                              hipStream_t stream) {
  (void)in_sizes; (void)n_in; (void)out_size; (void)ws_size;

  const float* x  = (const float*)d_in[0];  // [128,512,16,16]
  const float* w1 = (const float*)d_in[1];  // [256,512,3,3]
  const float* b1 = (const float*)d_in[2];  // [256]
  const float* w2 = (const float*)d_in[3];  // [128,256,3,3]
  const float* b2 = (const float*)d_in[4];  // [128]
  const float* w3 = (const float*)d_in[5];  // [1,128,1,1]
  const float* b3 = (const float*)d_in[6];  // [1]
  float* out = (float*)d_out;               // [128,1,512,512]

  // Workspace layout (~62 MB)
  char* ws = (char*)d_ws;
  _Float16* xb   = (_Float16*)(ws);               // 33,554,432 B : [128][256][512]
  _Float16* wp1  = (_Float16*)(ws + 33554432);    //  2,359,296 B : [9][512][256]
  _Float16* wp2  = (_Float16*)(ws + 35913728);    //    589,824 B : [9][256][128]
  _Float16* act1 = (_Float16*)(ws + 36503552);    // 16,777,216 B : [128][256][256]
  _Float16* act2 = (_Float16*)(ws + 53280768);    //  8,388,608 B : [128][256][128]
  float*    mask = (float*)   (ws + 61669376);    //    131,072 B : [128][256]

  // 1) NCHW f32 -> NHWC f16 (LDS transpose tiles)
  nchw_to_nhwc_f16<<<dim3(8, 16, 128), dim3(32, 8), 0, stream>>>(x, xb, 512);
  // 2) weight repacks
  pack_w<<<4608, 256, 0, stream>>>(w1, wp1, 256, 512);   // 1,179,648 elems
  pack_w<<<1152, 256, 0, stream>>>(w2, wp2, 128, 256);   //   294,912 elems
  // 3) conv1: 512 -> 256
  conv_wmma<<<dim3(4, 128), 256, 0, stream>>>(xb, wp1, b1, act1, 512, 256);
  // 4) conv2: 256 -> 128
  conv_wmma<<<dim3(2, 128), 256, 0, stream>>>(act1, wp2, b2, act2, 256, 128);
  // 5) 1x1 conv + top-5 + mask
  conv3_topk_mask<<<128, 256, 0, stream>>>(act2, w3, b3, mask);
  // 6) 32x upsample
  upsample_kernel<<<32768, 256, 0, stream>>>(mask, out);
}
